// Rnn_2860448219793
// MI455X (gfx1250) — matrix operand order repaired and checked by reading, not compiled, not run on hardware
//
#include <hip/hip_runtime.h>

// RNN (Elman) forward for MI455X / gfx1250.
// One wave32 owns 16 batch rows; h lives in registers across all T steps.
// Per step: D[hidden(2x16), batch16] = W_hh(f16 A, invariant) x h(f16 B)
//           + (x*W_ih + b)(f32 C) via two v_wmma_f32_16x16x32_f16,
// then hardware v_tanh_f32 and a LANE-LOCAL repack into the next-step B:
// the K-slots of A/B are permuted by kappa(k) (swap bit3<->bit4) so that the
// C/D lane layout of tanh(D) coincides with the B lane layout -- no
// ds_swizzle / cndmask on the recurrence critical path at all.
// Timesteps run in chunks of 4: one b128 x-load per chunk (prefetched one
// chunk ahead) and one predicated b128 output store per chunk. The only DS
// op per step is the lane^16 fold of the scalar output projection, which
// feeds only the store (off the critical path).

typedef __attribute__((ext_vector_type(16))) _Float16 v16h;
typedef __attribute__((ext_vector_type(2)))  _Float16 v2h;
typedef __attribute__((ext_vector_type(8)))  float    v8f;
typedef __attribute__((ext_vector_type(4)))  float    v4f;
typedef __attribute__((ext_vector_type(8)))  int      v8i;

__device__ __forceinline__ int swzi(int x) {
    // ds_swizzle_b32 group-of-32: and=0x1f, or=0, xor=0x10 -> lane ^ 16
    return __builtin_amdgcn_ds_swizzle(x, 0x401f);
}
__device__ __forceinline__ float swzf(float x) {
    return __int_as_float(swzi(__float_as_int(x)));
}

__device__ __forceinline__ float fast_tanh(float x) {
#if __has_builtin(__builtin_amdgcn_tanhf)
    return __builtin_amdgcn_tanhf(x);          // v_tanh_f32 (TRANS pipe)
#else
    float r;
    asm("v_tanh_f32 %0, %1\n\tv_nop" : "=v"(r) : "v"(x));
    return r;
#endif
}

__device__ __forceinline__ int pack2(float lo, float hi) {
    v2h p;                                     // v_cvt_pk_f16_f32 (RNE)
    p.x = (_Float16)lo;
    p.y = (_Float16)hi;
    return __builtin_bit_cast(int, p);
}

// K-slot -> hidden-unit permutation: swap bit3 and bit4.
// Blocks of 8: 0->0, 1->2, 2->1, 3->3. kappa is its own inverse.
__device__ __forceinline__ int kappa(int k) {
    return (k & 7) | ((k & 8) << 1) | ((k & 16) >> 1);
}

__global__ void __launch_bounds__(32)
rnn_wmma_kernel(const float* __restrict__ x,
                const float* __restrict__ h0,
                const float* __restrict__ W_ih,
                const float* __restrict__ b_ih,
                const float* __restrict__ W_hh,
                const float* __restrict__ b_hh,
                const float* __restrict__ W_out,
                const float* __restrict__ b_out,
                float* __restrict__ out,
                float* __restrict__ hT,
                int T)
{
    const int lane  = threadIdx.x;   // 0..31 (wave32)
    const int nloc  = lane & 15;     // batch column within tile / A row
    const int half  = lane >> 4;     // lane-half selects K/M ranges
    const int brow  = blockIdx.x * 16 + nloc;

    // ---- loop-invariant per-lane constants (C/D layout: m = v + 8*half) ----
    // M-slots are TRUE hidden indices (identity), so C / W_out / hT need no
    // permutation handling.
    float w0c[8], w1c[8], bias0[8], bias1[8], wo0[8], wo1[8];
#pragma unroll
    for (int v = 0; v < 8; ++v) {
        int m0 = v + 8 * half;       // tile0 hidden index
        int m1 = m0 + 16;            // tile1 hidden index
        w0c[v]   = W_ih[m0];
        w1c[v]   = W_ih[m1];
        bias0[v] = b_ih[m0] + b_hh[m0];
        bias1[v] = b_ih[m1] + b_hh[m1];
        wo0[v]   = W_out[m0];
        wo1[v]   = W_out[m1];
    }
    const float bO = b_out[0];

    // ---- A matrices: W_hh in f16 16x32 A-layout, columns kappa-permuted ----
    // lane l holds row M = l&15; VGPR j: K-slot = 2j (+8 if lane>=16) for j<4,
    //                                   K-slot = 16+2(j-4) (+8 if >=16) else.
    // A[m][k] = W_hh[m][k]  (K-slots are true hidden units in table order).
    v16h a0, a1;
#pragma unroll
    for (int j = 0; j < 8; ++j) {
        int jj = (j < 4) ? (2 * j) : (16 + 2 * (j - 4));
        int k  = jj + 8 * half;
        a0[2 * j]     = (_Float16)W_hh[nloc * 32 + k];
        a0[2 * j + 1] = (_Float16)W_hh[nloc * 32 + k + 1];
        a1[2 * j]     = (_Float16)W_hh[(nloc + 16) * 32 + k];
        a1[2 * j + 1] = (_Float16)W_hh[(nloc + 16) * 32 + k + 1];
    }

    // ---- initial B = h0 in f16 32x16 B-layout (K-slots kappa-permuted) -----
    // dword v, lane l: K-slot = 2v,2v+1 (lanes 0-15) or 16+2v,17+2v (16-31)
    v16h bm;
#pragma unroll
    for (int v = 0; v < 8; ++v) {
        int ku = kappa(2 * v + 16 * half);
        bm[2 * v]     = (_Float16)h0[brow * 32 + ku];
        bm[2 * v + 1] = (_Float16)h0[brow * 32 + ku + 1];
    }

    const float* xrow = x   + (size_t)brow * T;
    float*       orow = out + (size_t)brow * T;

    v8f d0 = {}, d1 = {};

    // One recurrence step: h_new = tanh(W_hh @ h + x*W_ih + b).
    // Updates bm/d0/d1, returns this step's output scalar (complete in ALL
    // lanes; caller stores from lanes 0-15).
    auto step = [&](float xv) -> float {
        v8f c0, c1;
#pragma unroll
        for (int v = 0; v < 8; ++v) {
            c0[v] = fmaf(xv, w0c[v], bias0[v]);
            c1[v] = fmaf(xv, w1c[v], bias1[v]);
        }
        d0 = __builtin_amdgcn_wmma_f32_16x16x32_f16(false, a0, false, bm,
                                                    (short)0, c0, false, false);
        d1 = __builtin_amdgcn_wmma_f32_16x16x32_f16(false, a1, false, bm,
                                                    (short)0, c1, false, false);
#pragma unroll
        for (int v = 0; v < 8; ++v) { d0[v] = fast_tanh(d0[v]); }
#pragma unroll
        for (int v = 0; v < 8; ++v) { d1[v] = fast_tanh(d1[v]); }

        // Lane-local repack of tanh(D) into the f16 B-layout. Thanks to the
        // kappa K-slot permutation this is identical for both lane halves:
        //   half 0 produces units {0..7}(d0) u {16..23}(d1)
        //          = units of K-slots 0..15   (kappa: 2v,2v+1 | 16+2v,17+2v)
        //   half 1 produces units {8..15}(d0) u {24..31}(d1)
        //          = units of K-slots 16..31
        v8i bi;
#pragma unroll
        for (int v = 0; v < 4; ++v) {
            bi[v]     = pack2(d0[2 * v], d0[2 * v + 1]);
            bi[v + 4] = pack2(d1[2 * v], d1[2 * v + 1]);
        }
        bm = __builtin_bit_cast(v16h, bi);

        // output projection: two independent fma chains, then lane^16 fold
        // (only feeds the store -- off the recurrence critical path)
        float pa = 0.f, pb = 0.f;
#pragma unroll
        for (int v = 0; v < 8; ++v) {
            pa = fmaf(d0[v], wo0[v], pa);
            pb = fmaf(d1[v], wo1[v], pb);
        }
        float p = pa + pb;
        p += swzf(p);
        return p + bO;
    };

    // ---- main loop: chunks of 4 timesteps, x prefetched one chunk ahead ----
    const int nchunks = T >> 2;
    if (nchunks > 0) {
        v4f cur = *(const v4f*)(xrow);
        for (int tq = 0; tq < nchunks; ++tq) {
            int tn = (tq + 1 < nchunks) ? (tq + 1) : tq;
            v4f nxt = *(const v4f*)(xrow + 4 * tn);   // issued early, waited
                                                      // only at `cur = nxt`
            v4f o;
            o.x = step(cur.x);
            o.y = step(cur.y);
            o.z = step(cur.z);
            o.w = step(cur.w);
            if (half == 0) *(v4f*)(orow + 4 * tq) = o;
            cur = nxt;
        }
    }
    for (int t = nchunks << 2; t < T; ++t) {          // tail (T % 4 != 0)
        float o = step(xrow[t]);
        if (half == 0) orow[t] = o;
    }

    // ---- final hidden state hT[b,k] (f32; M-slots are true indices) --------
#pragma unroll
    for (int v = 0; v < 8; ++v) {
        int k0 = v + 8 * half;
        hT[brow * 32 + k0]      = d0[v];
        hT[brow * 32 + k0 + 16] = d1[v];
    }
}

extern "C" void kernel_launch(void* const* d_in, const int* in_sizes, int n_in,
                              void* d_out, int out_size, void* d_ws, size_t ws_size,
                              hipStream_t stream) {
    const float* x     = (const float*)d_in[0];
    const float* h0    = (const float*)d_in[1];
    const float* W_ih  = (const float*)d_in[2];
    const float* b_ih  = (const float*)d_in[3];
    const float* W_hh  = (const float*)d_in[4];
    const float* b_hh  = (const float*)d_in[5];
    const float* W_out = (const float*)d_in[6];
    const float* b_out = (const float*)d_in[7];

    const int B = in_sizes[1] / 32;   // h_state has B*H elements, H=32
    const int T = in_sizes[0] / B;    // x has B*T*1 elements

    float* out = (float*)d_out;            // [B,T,1] flat
    float* hT  = out + (size_t)B * T;      // [1,B,32] flat, concatenated after

    dim3 grid(B / 16), block(32);          // one wave32 per 16 batch rows
    rnn_wmma_kernel<<<grid, block, 0, stream>>>(x, h0, W_ih, b_ih, W_hh, b_hh,
                                                W_out, b_out, out, hT, T);
}
